// CrossAttentionBlock_70403103916339
// MI455X (gfx1250) — matrix operand order repaired and checked by reading, not compiled, not run on hardware
//
#include <hip/hip_runtime.h>

#define E_DIM 256
#define L_SEQ 8192
#define B_SZ  4

typedef __attribute__((ext_vector_type(16))) __bf16 v16bf;
typedef __attribute__((ext_vector_type(8)))  __bf16 v8bf;
typedef __attribute__((ext_vector_type(8)))  float  v8f;

static __device__ __forceinline__ v16bf bcat(v8bf a, v8bf b) {
  return __builtin_shufflevector(a, b, 0,1,2,3,4,5,6,7,8,9,10,11,12,13,14,15);
}
static __device__ __forceinline__ v8f wmma_bf16(v16bf a, v16bf b, v8f c) {
  return __builtin_amdgcn_wmma_f32_16x16x32_bf16(false, a, false, b, (short)0, c, false, false);
}

// ---------------------------------------------------------------------------
// GEMM: Y[b,o,l] = scale * (sum_c W[o,c] X[b,c,l] + bias[o]) (+ R[b,o,l])
// bf16x3 split (AhBh + AhBl + AlBh) for ~fp32 accuracy on WMMA.
// Block tile 64(M) x 64(N), 8 waves each 16x32, K chunked by 64.
// ---------------------------------------------------------------------------
__global__ __launch_bounds__(256) void gemm256(
    const float* __restrict__ W, const float* __restrict__ X,
    const float* __restrict__ bias, const float* __restrict__ R,
    float* __restrict__ outF, __bf16* __restrict__ outB, float scale)
{
  __shared__ __align__(16) __bf16 WH[64][72];
  __shared__ __align__(16) __bf16 WL[64][72];
  __shared__ __align__(16) __bf16 XH[64][72];   // transposed: [n][k]
  __shared__ __align__(16) __bf16 XL[64][72];

  const int tid  = threadIdx.x;
  const int lane = tid & 31;
  const int wave = tid >> 5;
  const int n0 = blockIdx.x * 64;          // flat column base over B*L
  const int m0 = blockIdx.y * 64;          // output-channel base
  const int mo = (wave >> 1) * 16;
  const int no = (wave & 1) * 32;
  const int l15 = lane & 15;
  const int kh  = lane >> 4;

  v8f acc0 = {}; v8f acc1 = {};

  for (int kb = 0; kb < 4; ++kb) {
    __syncthreads();
    // stage W chunk [64 m][64 k] (row-major, float2 per thread)
    #pragma unroll
    for (int i = 0; i < 8; ++i) {
      int flat = i * 256 + tid;
      int m  = flat >> 5;
      int kk = (flat & 31) * 2;
      const float2 w2 = *reinterpret_cast<const float2*>(
          &W[(size_t)(m0 + m) * E_DIM + kb * 64 + kk]);
      __bf16 h0 = (__bf16)w2.x; __bf16 g0 = (__bf16)(w2.x - (float)h0);
      __bf16 h1 = (__bf16)w2.y; __bf16 g1 = (__bf16)(w2.y - (float)h1);
      WH[m][kk] = h0; WH[m][kk + 1] = h1;
      WL[m][kk] = g0; WL[m][kk + 1] = g1;
    }
    // stage X chunk transposed: XH[j][c]
    #pragma unroll
    for (int i = 0; i < 16; ++i) {
      int flat = i * 256 + tid;
      int c = flat >> 6;
      int j = flat & 63;
      int ng = n0 + j;
      int bb = ng >> 13;
      int ll = ng & (L_SEQ - 1);
      float xv = X[(size_t)bb * E_DIM * L_SEQ + (size_t)(kb * 64 + c) * L_SEQ + ll];
      __bf16 h = (__bf16)xv;
      XH[j][c] = h;
      XL[j][c] = (__bf16)(xv - (float)h);
    }
    if (kb < 3) {  // emits global_prefetch_b8
      __builtin_prefetch(&W[(size_t)(m0 + (tid >> 2)) * E_DIM + (kb + 1) * 64 + (tid & 3) * 16], 0, 1);
    }
    __syncthreads();
    #pragma unroll
    for (int ks = 0; ks < 2; ++ks) {
      // A operand: per lane row m=lane&15, 16 consecutive K at 16*kh
      const __bf16* pa = &WH[mo + l15][ks * 32 + 16 * kh];
      v16bf Ah = bcat(*(const v8bf*)pa, *(const v8bf*)(pa + 8));
      const __bf16* pal = &WL[mo + l15][ks * 32 + 16 * kh];
      v16bf Al = bcat(*(const v8bf*)pal, *(const v8bf*)(pal + 8));
      // B operand: per lane column n=lane&15, 16 consecutive K at 16*kh
      const __bf16* pb0  = &XH[no + l15][ks * 32 + 16 * kh];
      v16bf B0h = bcat(*(const v8bf*)pb0,  *(const v8bf*)(pb0 + 8));
      const __bf16* pb0l = &XL[no + l15][ks * 32 + 16 * kh];
      v16bf B0l = bcat(*(const v8bf*)pb0l, *(const v8bf*)(pb0l + 8));
      const __bf16* pb1  = &XH[no + 16 + l15][ks * 32 + 16 * kh];
      v16bf B1h = bcat(*(const v8bf*)pb1,  *(const v8bf*)(pb1 + 8));
      const __bf16* pb1l = &XL[no + 16 + l15][ks * 32 + 16 * kh];
      v16bf B1l = bcat(*(const v8bf*)pb1l, *(const v8bf*)(pb1l + 8));

      acc0 = wmma_bf16(Ah, B0h, acc0);
      acc0 = wmma_bf16(Ah, B0l, acc0);
      acc0 = wmma_bf16(Al, B0h, acc0);
      acc1 = wmma_bf16(Ah, B1h, acc1);
      acc1 = wmma_bf16(Ah, B1l, acc1);
      acc1 = wmma_bf16(Al, B1h, acc1);
    }
  }

  // epilogue: C/D layout row = r + 8*kh, col = lane&15
  {
    int ncol0 = n0 + no + l15;
    int bb = ncol0 >> 13;
    int ll = ncol0 & (L_SEQ - 1);
    #pragma unroll
    for (int r = 0; r < 8; ++r) {
      int o = m0 + mo + 8 * kh + r;
      float bv = bias[o];
      size_t idx0 = (size_t)bb * E_DIM * L_SEQ + (size_t)o * L_SEQ + ll;
      size_t idx1 = idx0 + 16;
      float v0 = (acc0[r] + bv) * scale;
      float v1 = (acc1[r] + bv) * scale;
      if (R) { v0 += R[idx0]; v1 += R[idx1]; }
      if (outF) { outF[idx0] = v0; outF[idx1] = v1; }
      else      { outB[idx0] = (__bf16)v0; outB[idx1] = (__bf16)v1; }
    }
  }
}

// ---------------------------------------------------------------------------
// Banded attention: per block 128 queries of one (b,h); key window 256.
// V staged with GLOBAL_LOAD_ASYNC_TO_LDS_B128 (ASYNCcnt) on interior blocks;
// K^T staged with a transpose (scalar). S = (Qhi+Qlo)*K^T, fp32 band mask +
// softmax, P relayout C->A via per-wave LDS (HW in-order, wave_barrier only).
// ---------------------------------------------------------------------------
__global__ __launch_bounds__(256) void banded_attn(
    const float* __restrict__ Q, const __bf16* __restrict__ K,
    const __bf16* __restrict__ V, float* __restrict__ Oa)
{
  __shared__ __align__(16) __bf16 KT[256][40];    // [key][d]
  __shared__ __align__(16) __bf16 Vs[32][264];    // [d][key]
  __shared__ __align__(16) __bf16 Ps[8][16][40];  // per wave [q][j]

  const int tid = threadIdx.x;
  const int lane = tid & 31;
  const int wave = tid >> 5;
  const int l15 = lane & 15;
  const int kh  = lane >> 4;
  const int bh = blockIdx.y;
  const int qb = blockIdx.x * 128;
  const int kw0 = qb - 64;
  const size_t base = (size_t)bh * 32 * L_SEQ;

  const bool interior = (kw0 >= 0) && (kw0 + 256 <= L_SEQ);

  // ---- V staging: [d][l] global -> [d][j] LDS (pure strided copy) ----
  if (interior) {
    // async DMA path: 1024 x 16B chunks, tracked by ASYNCcnt
    #pragma unroll
    for (int i = 0; i < 4; ++i) {
      int flat = i * 256 + tid;
      int d = flat >> 5;
      int j = (flat & 31) * 8;                 // element offset (16B chunks)
      unsigned ldsa = (unsigned)(uintptr_t)&Vs[d][j];
      unsigned long long ga =
          (unsigned long long)(uintptr_t)&V[base + (size_t)d * L_SEQ + kw0 + j];
      asm volatile("global_load_async_to_lds_b128 %0, %1, off"
                   :: "v"(ldsa), "v"(ga) : "memory");
    }
  } else {
    #pragma unroll
    for (int i = 0; i < 16; ++i) {
      int flat = i * 256 + tid;
      int d = flat >> 7;
      int j2 = (flat & 127) * 2;
      int l = kw0 + j2;
      __bf16 a = (__bf16)0.f, b = (__bf16)0.f;
      if (l >= 0 && l < L_SEQ)         a = V[base + (size_t)d * L_SEQ + l];
      if (l + 1 >= 0 && l + 1 < L_SEQ) b = V[base + (size_t)d * L_SEQ + l + 1];
      Vs[d][j2] = a; Vs[d][j2 + 1] = b;
    }
  }

  // ---- K^T staging (transpose, zero-fill out of range) ----
  #pragma unroll
  for (int i = 0; i < 32; ++i) {
    int flat = i * 256 + tid;
    int d = flat >> 8;
    int j = flat & 255;
    int l = kw0 + j;
    __bf16 kv = (__bf16)0.f;
    if (l >= 0 && l < L_SEQ) kv = K[base + (size_t)d * L_SEQ + l];
    KT[j][d] = kv;
  }

  if (interior) {
    asm volatile("s_wait_asynccnt 0x0" ::: "memory");
  }
  __syncthreads();

  // Q A-operand straight from global into registers, hi/lo split (scale pre-folded)
  const int qw = qb + wave * 16;
  const int qg = qw + l15;
  v16bf Ah, Al;
  #pragma unroll
  for (int e = 0; e < 8; ++e) {
    float f0 = Q[base + (size_t)(16 * kh + e) * L_SEQ + qg];
    __bf16 h0 = (__bf16)f0;
    Ah[e] = h0; Al[e] = (__bf16)(f0 - (float)h0);
    float f1 = Q[base + (size_t)(16 * kh + 8 + e) * L_SEQ + qg];
    __bf16 h1 = (__bf16)f1;
    Ah[8 + e] = h1; Al[8 + e] = (__bf16)(f1 - (float)h1);
  }

  v8f s[9];
  #pragma unroll
  for (int tt = 0; tt < 9; ++tt) {    // 9 key tiles cover the +-63 band
    int t = wave + tt;
    const __bf16* pk = &KT[t * 16 + l15][16 * kh];
    v16bf Bk = bcat(*(const v8bf*)pk, *(const v8bf*)(pk + 8));
    v8f z = {};
    z = wmma_bf16(Ah, Bk, z);
    z = wmma_bf16(Al, Bk, z);
    s[tt] = z;
  }

  v8f rmax, rsum;
  #pragma unroll
  for (int r = 0; r < 8; ++r) { rmax[r] = -1e30f; rsum[r] = 0.f; }

  #pragma unroll
  for (int tt = 0; tt < 9; ++tt) {    // band + sequence mask
    int kc = kw0 + (wave + tt) * 16 + l15;
    bool kin = (kc >= 0) && (kc < L_SEQ);
    #pragma unroll
    for (int r = 0; r < 8; ++r) {
      int qr = qw + 8 * kh + r;
      int dlt = qr - kc;
      bool ok = kin && (dlt < 64) && (dlt > -64);
      float v = ok ? s[tt][r] : -1e30f;
      s[tt][r] = v;
      rmax[r] = fmaxf(rmax[r], v);
    }
  }
  #pragma unroll
  for (int off = 8; off >= 1; off >>= 1) {
    #pragma unroll
    for (int r = 0; r < 8; ++r)
      rmax[r] = fmaxf(rmax[r], __shfl_xor(rmax[r], off, 32));
  }
  #pragma unroll
  for (int tt = 0; tt < 9; ++tt) {
    #pragma unroll
    for (int r = 0; r < 8; ++r) {
      float p = __expf(s[tt][r] - rmax[r]);
      s[tt][r] = p;
      rsum[r] += p;
    }
  }
  #pragma unroll
  for (int off = 8; off >= 1; off >>= 1) {
    #pragma unroll
    for (int r = 0; r < 8; ++r)
      rsum[r] += __shfl_xor(rsum[r], off, 32);
  }

  // PV: per-wave LDS relayout of P (C-layout -> A-layout). All traffic is
  // intra-wave and same-wave LDS ops are HW in-order -> only a compiler
  // scheduling fence is needed, no workgroup barrier.
  v8f o0 = {}; v8f o1 = {};
  const int par = wave & 1;
  const int tbase = wave - par;       // even tile base for 32-wide PV K-steps
  #pragma unroll
  for (int pp = 0; pp < 5; ++pp) {
    #pragma unroll
    for (int u = 0; u < 2; ++u) {
      const int ttA = pp * 2 + u;     // tile index rel. to this wave (par==0)
      const int ttB = ttA - 1;        // (par==1)
      #pragma unroll
      for (int r = 0; r < 8; ++r) {
        float pv = 0.f;
        if (par == 0) { if (ttA < 9) pv = s[ttA][r]; }
        else          { if (ttB >= 0) pv = s[ttB][r]; }
        Ps[wave][8 * kh + r][u * 16 + l15] = (__bf16)pv;
      }
    }
    __builtin_amdgcn_wave_barrier();
    const __bf16* pP = &Ps[wave][l15][16 * kh];
    v16bf Pa = bcat(*(const v8bf*)pP, *(const v8bf*)(pP + 8));
    int jb = (tbase + pp * 2) * 16;
    const __bf16* pv0 = &Vs[l15][jb + 16 * kh];
    v16bf Bv0 = bcat(*(const v8bf*)pv0, *(const v8bf*)(pv0 + 8));
    const __bf16* pv1 = &Vs[16 + l15][jb + 16 * kh];
    v16bf Bv1 = bcat(*(const v8bf*)pv1, *(const v8bf*)(pv1 + 8));
    o0 = wmma_bf16(Pa, Bv0, o0);
    o1 = wmma_bf16(Pa, Bv1, o1);
    __builtin_amdgcn_wave_barrier();
  }

  #pragma unroll
  for (int r = 0; r < 8; ++r) {
    float inv = 1.f / rsum[r];
    int qr = qw + 8 * kh + r;
    Oa[base + (size_t)l15 * L_SEQ + qr]        = o0[r] * inv;
    Oa[base + (size_t)(16 + l15) * L_SEQ + qr] = o1[r] * inv;
  }
}

// ---------------------------------------------------------------------------
extern "C" void kernel_launch(void* const* d_in, const int* in_sizes, int n_in,
                              void* d_out, int out_size, void* d_ws, size_t ws_size,
                              hipStream_t stream) {
  (void)in_sizes; (void)n_in; (void)out_size; (void)ws_size;
  const float* x  = (const float*)d_in[0];
  const float* Wq = (const float*)d_in[1];
  const float* bq = (const float*)d_in[2];
  const float* Wk = (const float*)d_in[3];
  const float* bk = (const float*)d_in[4];
  const float* Wv = (const float*)d_in[5];
  const float* bv = (const float*)d_in[6];
  const float* Wo = (const float*)d_in[7];
  const float* bo = (const float*)d_in[8];

  char* ws = (char*)d_ws;
  float*  Qb = (float*)ws;                            // 32 MB fp32
  __bf16* Kb = (__bf16*)(ws + (size_t)33554432);      // 16 MB bf16
  __bf16* Vb = (__bf16*)(ws + (size_t)50331648);      // 16 MB bf16
  float*  Ab = (float*)(ws + (size_t)67108864);       // 32 MB fp32

  dim3 blk(256);
  dim3 gg(B_SZ * L_SEQ / 64, E_DIM / 64);             // (512, 4)
  const float qscale = 0.17677669529663687f;          // 1/sqrt(32)

  gemm256<<<gg, blk, 0, stream>>>(Wq, x, bq, nullptr, Qb, nullptr, qscale);
  gemm256<<<gg, blk, 0, stream>>>(Wk, x, bk, nullptr, nullptr, Kb, 1.0f);
  gemm256<<<gg, blk, 0, stream>>>(Wv, x, bv, nullptr, nullptr, Vb, 1.0f);

  dim3 ga(L_SEQ / 128, B_SZ * 8);                     // (64, 32)
  banded_attn<<<ga, blk, 0, stream>>>(Qb, Kb, Vb, Ab);

  gemm256<<<gg, blk, 0, stream>>>(Wo, Ab, bo, x, (float*)d_out, nullptr, 1.0f);
}
